// HybridContrastiveLoss_47364899340368
// MI455X (gfx1250) — matrix operand order repaired and checked by reading, not compiled, not run on hardware
//
#include <hip/hip_runtime.h>
#include <hip/hip_fp16.h>

// ---------------------------------------------------------------------------
// HybridContrastiveLoss on MI455X (gfx1250, wave32)
//   loss_pixel : fused f16 WMMA Gram-matrix, split-K over 8 column chunks
//                (4096 waves) + per-row partial reduce kernel
//   loss_local : wave-per-pixel, lanes stripe the 11x11 neighborhood
//   loss_dir   : 2-direction contrastive, f32 VALU
// ---------------------------------------------------------------------------

#define TEMP_INV 10.0f       // 1 / 0.1
#define EPSF     1e-6f
#define NBHD     5

typedef __attribute__((ext_vector_type(16))) _Float16 v16h;
typedef __attribute__((ext_vector_type(8)))  float    v8f;

static constexpr int N_  = 2;
static constexpr int C_  = 64;
static constexpr int H_  = 64;
static constexpr int W_  = 64;
static constexpr int HW_ = H_ * W_;      // 4096
static constexpr int M_  = N_ * HW_;     // 8192 pixels
static constexpr int JT_ = M_ / 16;      // 512 column tiles
static constexpr int JC_ = 8;            // split-K chunks
static constexpr int JPC_ = JT_ / JC_;   // 64 tiles per chunk

union FragAB {
    v16h    h;
    uint4   q[2];
};

// --------------------------------------------------------------------------
__global__ void k_init(float* out) {
    if (threadIdx.x == 0 && blockIdx.x == 0) out[0] = 0.0f;
}

// --------------------------------------------------------------------------
// Normalize features (N,C,H,W) -> fm32[p][c], fm16[p][c] with p=(n,h,w)
__global__ void k_normalize(const float* __restrict__ feat,
                            float* __restrict__ fm32,
                            _Float16* __restrict__ fm16) {
    int p = blockIdx.x * blockDim.x + threadIdx.x;
    if (p >= M_) return;
    int n  = p / HW_;
    int hw = p % HW_;
    const float* src = feat + (size_t)n * C_ * HW_ + hw;

    float v[C_];
    float ss = 0.0f;
#pragma unroll
    for (int c = 0; c < C_; ++c) {
        float x = src[(size_t)c * HW_];
        v[c] = x;
        ss += x * x;
    }
    float inv = 1.0f / fmaxf(sqrtf(ss), 1e-12f);

    float*    d32 = fm32 + (size_t)p * C_;
    _Float16* d16 = fm16 + (size_t)p * C_;
#pragma unroll
    for (int c = 0; c < C_; ++c) {
        float y = v[c] * inv;
        d32[c] = y;
        d16[c] = (_Float16)y;
    }
}

// --------------------------------------------------------------------------
// loss_pixel stage 1: wave g handles row-tile (g & 511) x column-chunk
// (g >> 9).  Inner loop: two v_wmma_f32_16x16x32_f16 per 16x16 tile (K=64),
// online accumulation of den = sum exp(l)*mask and L = sum (mask ? l : -1e30).
// Per-row partials written to denP/LP workspace.
__global__ __launch_bounds__(256)
void k_pixel_loss(const _Float16* __restrict__ fm16,
                  const int* __restrict__ lab,
                  float* __restrict__ denP,
                  float* __restrict__ LP) {
    const int lane  = threadIdx.x & 31;
    const int wave  = threadIdx.x >> 5;
    const int g     = blockIdx.x * 8 + wave;    // 0..4095
    const int tile  = g & (JT_ - 1);            // 0..511
    const int chunk = g >> 9;                   // 0..7
    const int i_base = tile * 16;
    const int hl   = lane >> 4;                 // half-wave (0/1)
    const int nsub = lane & 15;

    // A fragments (16x32 f16, two k-steps): two contiguous 16B spans / lane.
    FragAB a0, a1;
    {
        const uint4* row = (const uint4*)(fm16 + (size_t)(i_base + nsub) * C_);
        a0.q[0] = row[hl];
        a0.q[1] = row[2 + hl];
        a1.q[0] = row[4 + hl];
        a1.q[1] = row[6 + hl];
    }

    int labI[8];
#pragma unroll
    for (int r = 0; r < 8; ++r) labI[r] = lab[i_base + r + 8 * hl];

    float den[8], Ls[8];
#pragma unroll
    for (int r = 0; r < 8; ++r) { den[r] = 0.0f; Ls[r] = 0.0f; }

    const int jt0 = chunk * JPC_;
    for (int jt = jt0; jt < jt0 + JPC_; ++jt) {
        const int j_base = jt * 16;
        // B fragments (32x16 f16): two 16B spans per k-step / lane (same k order as A).
        FragAB b0, b1;
        const uint4* rowj = (const uint4*)(fm16 + (size_t)(j_base + nsub) * C_);
        b0.q[0] = rowj[hl];
        b0.q[1] = rowj[2 + hl];
        b1.q[0] = rowj[4 + hl];
        b1.q[1] = rowj[6 + hl];

        v8f acc = {0.f, 0.f, 0.f, 0.f, 0.f, 0.f, 0.f, 0.f};
        acc = __builtin_amdgcn_wmma_f32_16x16x32_f16(
                  false, a0.h, false, b0.h, (short)0, acc, false, false);
        acc = __builtin_amdgcn_wmma_f32_16x16x32_f16(
                  false, a1.h, false, b1.h, (short)0, acc, false, false);

        const int labJ = lab[j_base + nsub];
#pragma unroll
        for (int r = 0; r < 8; ++r) {
            float l = acc[r] * TEMP_INV;
            bool mk = (labI[r] == labJ);
            den[r] += mk ? __expf(l) : 0.0f;
            Ls[r]  += mk ? l : -1e30f;
        }
    }

    // Reduce across the 16 lanes sharing each row (xor stays within halves).
#pragma unroll
    for (int r = 0; r < 8; ++r) {
        float d = den[r], L = Ls[r];
#pragma unroll
        for (int m = 1; m < 16; m <<= 1) {
            d += __shfl_xor(d, m, 32);
            L += __shfl_xor(L, m, 32);
        }
        den[r] = d; Ls[r] = L;
    }

    if (nsub == 0) {
#pragma unroll
        for (int r = 0; r < 8; ++r) {
            int row = i_base + r + 8 * hl;
            denP[(size_t)chunk * M_ + row] = den[r];
            LP  [(size_t)chunk * M_ + row] = Ls[r];
        }
    }
}

// --------------------------------------------------------------------------
// loss_pixel stage 2: one thread per row; combine split-K partials.
__global__ void k_pixel_reduce(const float* __restrict__ denP,
                               const float* __restrict__ LP,
                               float* __restrict__ out) {
    int row = blockIdx.x * blockDim.x + threadIdx.x;
    if (row >= M_) return;
    float den = 0.0f, L = 0.0f;
#pragma unroll
    for (int c = 0; c < JC_; ++c) {
        den += denP[(size_t)c * M_ + row];
        L   += LP  [(size_t)c * M_ + row];
    }
    float contrib = (float)M_ * __logf(den + EPSF) - L;
    atomicAdd(out, contrib * (1.0f / ((float)M_ * (float)M_)));
}

// --------------------------------------------------------------------------
// loss_local: one WAVE per (n,h,w); lanes stripe over the <=121 neighbors,
// then a wave reduction combines den / Lsum.
__global__ __launch_bounds__(256)
void k_local_loss(const float* __restrict__ fm32,
                  const int* __restrict__ lab,
                  float* __restrict__ out) {
    const int lane = threadIdx.x & 31;
    const int wave = threadIdx.x >> 5;
    const int p = blockIdx.x * 8 + wave;        // one wave per pixel
    int n  = p / HW_;
    int hw = p % HW_;
    int h  = hw / W_, w = hw % W_;

    const int h0 = max(h - NBHD, 0), h1 = min(h + NBHD, H_ - 1);
    const int w0 = max(w - NBHD, 0), w1 = min(w + NBHD, W_ - 1);
    const int Kw  = (w1 - w0 + 1);
    const int cnt = (h1 - h0 + 1) * Kw;

    const float4* me = (const float4*)(fm32 + (size_t)p * C_);
    const int labMe = lab[p];

    float den = 0.0f, Lsum = 0.0f;
    for (int idx = lane; idx < cnt; idx += 32) {
        int nh = h0 + idx / Kw;
        int nw = w0 + idx % Kw;
        int q = n * HW_ + nh * W_ + nw;
        const float4* nb = (const float4*)(fm32 + (size_t)q * C_);
        float dot = 0.0f;
#pragma unroll
        for (int t = 0; t < 16; ++t) {
            float4 a = me[t];
            float4 b = nb[t];
            dot += a.x * b.x + a.y * b.y + a.z * b.z + a.w * b.w;
        }
        float l = dot * TEMP_INV;
        bool mk = (lab[q] == labMe);
        den  += mk ? __expf(l) : 0.0f;
        Lsum += mk ? l : -1e30f;
    }

#pragma unroll
    for (int m = 1; m < 32; m <<= 1) {
        den  += __shfl_xor(den, m, 32);
        Lsum += __shfl_xor(Lsum, m, 32);
    }

    if (lane == 0) {
        float term = (float)cnt * __logf(den + EPSF) - Lsum;
        atomicAdd(out, term / ((float)N_ * (float)cnt * (float)HW_));
    }
}

// --------------------------------------------------------------------------
// loss_dir: one thread per (n,h,w); 2 per-pixel integer-offset neighbors.
__global__ void k_dir_loss(const float* __restrict__ fm32,
                           const int* __restrict__ lab,
                           const float* __restrict__ dirs,
                           float* __restrict__ out) {
    int p = blockIdx.x * blockDim.x + threadIdx.x;
    if (p >= M_) return;
    int n  = p / HW_;
    int hw = p % HW_;
    int h  = hw / W_, w = hw % W_;

    const float4* me = (const float4*)(fm32 + (size_t)p * C_);
    const int labMe = lab[p];

    float lg[2];
    bool  vld[2], lm[2];
    float den = 0.0f;
    int   kc  = 0;
#pragma unroll
    for (int k = 0; k < 2; ++k) {
        int di = (int)dirs[((size_t)k * 2 + 0) * HW_ + hw];  // trunc toward 0
        int dj = (int)dirs[((size_t)k * 2 + 1) * HW_ + hw];
        int ni = h + di, nj = w + dj;
        bool v = (ni >= 0) && (ni < H_) && (nj >= 0) && (nj < W_);
        int nic = min(max(ni, 0), H_ - 1);
        int njc = min(max(nj, 0), W_ - 1);
        int q = n * HW_ + nic * W_ + njc;
        const float4* nb = (const float4*)(fm32 + (size_t)q * C_);
        float dot = 0.0f;
#pragma unroll
        for (int t = 0; t < 16; ++t) {
            float4 a = me[t];
            float4 b = nb[t];
            dot += a.x * b.x + a.y * b.y + a.z * b.z + a.w * b.w;
        }
        float l = dot * TEMP_INV;
        bool m = (lab[q] == labMe);
        lg[k] = l; vld[k] = v; lm[k] = m;
        den += (v && m) ? __expf(l) : 0.0f;
        kc  += v ? 1 : 0;
    }
    if (kc > 0) {
        float ld = __logf(den + EPSF);
        float termsum = 0.0f;
#pragma unroll
        for (int k = 0; k < 2; ++k)
            if (vld[k]) termsum += ld - (lm[k] ? lg[k] : -1e30f);
        atomicAdd(out, termsum / ((float)N_ * (float)kc * (float)HW_));
    }
}

// --------------------------------------------------------------------------
extern "C" void kernel_launch(void* const* d_in, const int* in_sizes, int n_in,
                              void* d_out, int out_size, void* d_ws, size_t ws_size,
                              hipStream_t stream) {
    (void)in_sizes; (void)n_in; (void)out_size; (void)ws_size;
    const float* feat = (const float*)d_in[0];
    const int*   lab  = (const int*)d_in[1];
    const float* dirs = (const float*)d_in[2];
    float*       out  = (float*)d_out;

    char* ws = (char*)d_ws;
    float*    fm32 = (float*)ws;                                    // 2 MB
    _Float16* fm16 = (_Float16*)(ws + (size_t)M_ * C_ * 4);         // 1 MB
    float*    denP = (float*)(ws + (size_t)M_ * C_ * 6);            // 256 KB
    float*    LP   = (float*)(ws + (size_t)M_ * C_ * 6
                                 + (size_t)JC_ * M_ * 4);           // 256 KB

    k_init<<<1, 1, 0, stream>>>(out);
    k_normalize<<<M_ / 256, 256, 0, stream>>>(feat, fm32, fm16);
    k_pixel_loss<<<(JT_ * JC_) / 8, 256, 0, stream>>>(fm16, lab, denP, LP);
    k_pixel_reduce<<<M_ / 256, 256, 0, stream>>>(denP, LP, out);
    k_local_loss<<<M_ / 8, 256, 0, stream>>>(fm32, lab, out);
    k_dir_loss<<<M_ / 256, 256, 0, stream>>>(fm32, lab, dirs, out);
}
